// V2DensePacketTokenizer_44427141710616
// MI455X (gfx1250) — matrix operand order repaired and checked by reading, compiled, not run on hardware
//
#include <hip/hip_runtime.h>
#include <cmath>

// ---------------------------------------------------------------------------
// V2DensePacketTokenizer for MI455X (gfx1250, wave32, WMMA bf16 16x16x32)
//
// reference: x = sign(f)*log1p(min(|sanitize(f)|,1e8)) ++ mask, pad to 4096,
//            h[b,t,:] = chunk[b,t,:] @ W[t] + b[t];  h = SiLU(h);
//            out = LayerNorm_512(h) * gamma[t] + beta[t]
//
// B=4096, T=32, C=128, D=512. Output-write bound (268 MB) -> bf16 WMMA.
// ---------------------------------------------------------------------------

#define T_BATCH   4096
#define T_INDIM   2047
#define T_DMODEL  512
#define T_TOKENS  32
#define T_CHUNK   128
#define BM        32          // batch rows per block (2 m-tiles)
#define LDS_AST   136         // padded A stride (bf16 elems) to avoid bank conflicts

typedef __attribute__((ext_vector_type(16))) __bf16 bf16x16;
typedef __attribute__((ext_vector_type(8)))  float  f32x8;

union FragAB {
    unsigned int u[8];
    bf16x16      b;
};

__device__ __forceinline__ unsigned short f2bf(float x) {
    unsigned int u = __float_as_uint(x);
    // round-to-nearest-even truncation to bf16 (inputs are sanitized/finite)
    return (unsigned short)((u + 0x7FFFu + ((u >> 16) & 1u)) >> 16);
}

// ---------------------------------------------------------------------------
// Pack W[t][c][d] (f32) into bf16 fragments in the per-lane order this kernel uses for the
// WMMA B operand (contiguous k, matched by A): frag element i of lane L  <->  K = 16*(L>>4)+i,
// N = L&15.  Layout: Wp[((t*4+ks)*32+nt)*512 + L*16 + i].
// Each thread writes one packed dword (two consecutive i).
// ---------------------------------------------------------------------------
__global__ __launch_bounds__(256) void pack_w_kernel(const float* __restrict__ W,
                                                     unsigned short* __restrict__ Wp) {
    int p  = blockIdx.x * 256 + threadIdx.x;   // pair index, 1,048,576 total
    int i2 = p * 2;
    int i    =  i2        & 15;
    int lane = (i2 >> 4)  & 31;
    int nt   = (i2 >> 9)  & 31;
    int ks   = (i2 >> 14) & 3;
    int t    =  i2 >> 16;
    int d = nt * 16 + (lane & 15);
    int c = ks * 32 + 16 * (lane >> 4) + i;
    unsigned int lo = f2bf(W[((size_t)(t * T_CHUNK + c)     ) * T_DMODEL + d]);
    unsigned int hi = f2bf(W[((size_t)(t * T_CHUNK + c + 1) ) * T_DMODEL + d]);
    *(unsigned int*)(Wp + i2) = (hi << 16) | lo;
}

// ---------------------------------------------------------------------------
// Fused tokenize + grouped GEMM + bias + SiLU + LayerNorm.
// grid = (4096/32 batch tiles, 32 tokens), block = 256 threads (8 wave32).
// Wave w owns output columns [w*64, w*64+64): 4 n-tiles; 2 m-tiles; K=128.
// ---------------------------------------------------------------------------
__global__ __launch_bounds__(256) void tokenizer_gemm_kernel(
        const float* __restrict__ features,
        const float* __restrict__ missing_mask,
        const unsigned short* __restrict__ Wp,
        const float* __restrict__ bias,
        const float* __restrict__ gamma,
        const float* __restrict__ beta,
        float* __restrict__ out) {

    __shared__ unsigned short As[BM][LDS_AST];   // bf16 A tile, padded stride
    __shared__ float rowsum[BM];
    __shared__ float rowsq[BM];

    const int tid  = threadIdx.x;
    const int lane = tid & 31;
    const int wave = tid >> 5;
    const int half = lane >> 4;    // which 16-lane half
    const int nlo  = lane & 15;
    const int m0   = blockIdx.x * BM;
    const int t    = blockIdx.y;

    if (tid < BM) { rowsum[tid] = 0.0f; rowsq[tid] = 0.0f; }

    // ---- stage A tile: sanitize + signed log1p / mask / zero pad -> bf16 LDS
    for (int e = tid; e < BM * T_CHUNK; e += 256) {
        int row = e >> 7;
        int col = e & 127;
        int g   = t * T_CHUNK + col;              // global effective-dim index
        long long b_ = (long long)(m0 + row);
        float v;
        if (g < T_INDIM) {
            float f = features[b_ * T_INDIM + g];
            if (__builtin_isnan(f) || __builtin_isinf(f)) f = 0.0f;
            v = copysignf(log1pf(fminf(fabsf(f), 1e8f)), f);
        } else if (g < 2 * T_INDIM) {
            v = missing_mask[b_ * T_INDIM + (g - T_INDIM)];
        } else {
            v = 0.0f;                             // 2-element pad
        }
        As[row][col] = f2bf(v);
    }
    __syncthreads();

    // ---- per-column params for this wave's 4 n-tiles
    float biasv[4], gmv[4], btv[4];
#pragma unroll
    for (int nt = 0; nt < 4; ++nt) {
        int d = wave * 64 + nt * 16 + nlo;
        biasv[nt] = bias [t * T_DMODEL + d];
        gmv[nt]   = gamma[t * T_DMODEL + d];
        btv[nt]   = beta [t * T_DMODEL + d];
    }

    f32x8 acc[2][4];
    const f32x8 zz = {0.f, 0.f, 0.f, 0.f, 0.f, 0.f, 0.f, 0.f};
#pragma unroll
    for (int mt = 0; mt < 2; ++mt)
#pragma unroll
        for (int nt = 0; nt < 4; ++nt) acc[mt][nt] = zz;

    const unsigned short* wbase = Wp + (size_t)t * (4 * 32 * T_DMODEL);

    // ---- K loop: 4 steps of 32
    for (int ks = 0; ks < 4; ++ks) {
        // A fragments (16-bit A layout: K=2v+16*half, contiguous to match packed B)
        FragAB afr[2];
#pragma unroll
        for (int mt = 0; mt < 2; ++mt) {
            const unsigned short* rp = &As[mt * 16 + nlo][ks * 32 + 16 * half];
#pragma unroll
            for (int v = 0; v < 8; ++v) {
                int off = 2 * v;
                afr[mt].u[v] = *(const unsigned int*)(rp + off);
            }
        }
#pragma unroll
        for (int nt = 0; nt < 4; ++nt) {
            // B fragment: 32 contiguous bytes per lane, pre-packed -> 2x b128
            const unsigned short* bp =
                wbase + ((size_t)(ks * 32 + (wave * 4 + nt)) * T_DMODEL) + lane * 16;
            const uint4* bp4 = (const uint4*)bp;
            uint4 q0 = bp4[0];
            uint4 q1 = bp4[1];
            FragAB bfr;
            bfr.u[0] = q0.x; bfr.u[1] = q0.y; bfr.u[2] = q0.z; bfr.u[3] = q0.w;
            bfr.u[4] = q1.x; bfr.u[5] = q1.y; bfr.u[6] = q1.z; bfr.u[7] = q1.w;
#pragma unroll
            for (int mt = 0; mt < 2; ++mt) {
                acc[mt][nt] = __builtin_amdgcn_wmma_f32_16x16x32_bf16(
                    /*neg_a=*/false, afr[mt].b,
                    /*neg_b=*/false, bfr.b,
                    /*c_mod=*/(short)0, acc[mt][nt],
                    /*reuse_a=*/false, /*reuse_b=*/false);
            }
        }
    }

    // ---- bias + SiLU (in accumulator registers)
#pragma unroll
    for (int mt = 0; mt < 2; ++mt)
#pragma unroll
        for (int nt = 0; nt < 4; ++nt)
#pragma unroll
            for (int r = 0; r < 8; ++r) {
                float x = acc[mt][nt][r] + biasv[nt];
                float h = x / (1.0f + __expf(-x));
                acc[mt][nt][r] = h;
            }

    // ---- LayerNorm reduction over D=512 (spread across 8 waves)
    // C layout: lane L, vgpr r -> row = r + 8*(L>>4), col = L&15
#pragma unroll
    for (int mt = 0; mt < 2; ++mt)
#pragma unroll
        for (int r = 0; r < 8; ++r) {
            float s = 0.0f, q = 0.0f;
#pragma unroll
            for (int nt = 0; nt < 4; ++nt) {
                float v = acc[mt][nt][r];
                s += v;
                q += v * v;
            }
            // butterfly across the 16 lanes holding the same row
#pragma unroll
            for (int m = 1; m <= 8; m <<= 1) {
                s += __shfl_xor(s, m, 32);
                q += __shfl_xor(q, m, 32);
            }
            if (nlo == 0) {
                int row = mt * 16 + r + 8 * half;
                atomicAdd(&rowsum[row], s);   // ds_add_f32
                atomicAdd(&rowsq[row], q);
            }
        }
    __syncthreads();

    // ---- normalize + affine + store
#pragma unroll
    for (int mt = 0; mt < 2; ++mt)
#pragma unroll
        for (int r = 0; r < 8; ++r) {
            int row = mt * 16 + r + 8 * half;
            float mean = rowsum[row] * (1.0f / 512.0f);
            float var  = rowsq[row] * (1.0f / 512.0f) - mean * mean;
            float rstd = rsqrtf(var + 1e-5f);
            size_t ob = ((size_t)(m0 + row) * T_TOKENS + t) * T_DMODEL;
#pragma unroll
            for (int nt = 0; nt < 4; ++nt) {
                int d = wave * 64 + nt * 16 + nlo;
                out[ob + d] = (acc[mt][nt][r] - mean) * rstd * gmv[nt] + btv[nt];
            }
        }
}

// ---------------------------------------------------------------------------
extern "C" void kernel_launch(void* const* d_in, const int* in_sizes, int n_in,
                              void* d_out, int out_size, void* d_ws, size_t ws_size,
                              hipStream_t stream) {
    (void)in_sizes; (void)n_in; (void)out_size; (void)ws_size;
    const float* features = (const float*)d_in[0];   // [4096, 2047] f32
    const float* mask     = (const float*)d_in[1];   // [4096, 2047] f32
    const float* W        = (const float*)d_in[2];   // [32, 128, 512] f32
    const float* bias     = (const float*)d_in[3];   // [32, 512] f32
    const float* gamma    = (const float*)d_in[4];   // [32, 512] f32
    const float* beta     = (const float*)d_in[5];   // [32, 512] f32
    float* out            = (float*)d_out;           // [4096, 32, 512] f32

    unsigned short* Wp = (unsigned short*)d_ws;      // 4 MB bf16 packed W

    // one-shot W repack into WMMA B-fragment order (bf16)
    pack_w_kernel<<<4096, 256, 0, stream>>>(W, Wp);

    dim3 grid(T_BATCH / BM, T_TOKENS);               // (128, 32)
    tokenizer_gemm_kernel<<<grid, 256, 0, stream>>>(features, mask, Wp,
                                                    bias, gamma, beta, out);
}
